// GraphNeuralReasoner_10806137717190
// MI455X (gfx1250) — matrix operand order repaired and checked by reading, not compiled, not run on hardware
//
#include <hip/hip_runtime.h>
#include <hip/hip_bf16.h>

typedef __attribute__((ext_vector_type(16))) _Float16 v16h;
typedef __attribute__((ext_vector_type(8)))  float    v8f;

static inline size_t align_up(size_t x, size_t a) { return (x + a - 1) & ~(a - 1); }
static inline int div_up_i(long long n, int d) { return (int)((n + d - 1) / d); }

// ---------------------------------------------------------------- utilities

__global__ void fill_kernel(float* __restrict__ p, float v, int n) {
  int i = blockIdx.x * blockDim.x + threadIdx.x;
  if (i < n) p[i] = v;
}

__global__ void deg_accum_kernel(const int* __restrict__ dst, float* __restrict__ deg, int n_edges) {
  int e = blockIdx.x * blockDim.x + threadIdx.x;
  if (e < n_edges) atomicAdd(&deg[dst[e]], 1.0f);
}

__global__ void rsqrt_inplace_kernel(float* __restrict__ p, int n) {
  int i = blockIdx.x * blockDim.x + threadIdx.x;
  if (i < n) p[i] = rsqrtf(p[i]);
}

// W [128x128] row-major (k,n) -> split f16 W^T: Wh[n][k] = fp16(w), Wl[n][k] = fp16(w - Wh)
__global__ void convert_w_kernel(const float* __restrict__ W,
                                 _Float16* __restrict__ Wh, _Float16* __restrict__ Wl) {
  int t = blockIdx.x * blockDim.x + threadIdx.x;   // 16384 threads
  int k = t >> 7;
  int n = t & 127;
  float w = W[k * 128 + n];
  _Float16 hi = (_Float16)w;
  Wh[n * 128 + k] = hi;
  Wl[n * 128 + k] = (_Float16)(w - (float)hi);
}

// ---------------------------------------------------------------- WMMA GEMM (split-f16, near-f32 accuracy)
// out[N,128] = act(A[N,128]) @ W[128,128]
// One wave -> one 16x16 tile; block = 8 waves = 16 rows x 128 cols; K=128 via 4 K-steps,
// 3 WMMAs per K-step: Ah*Wh + Al*Wh + Ah*Wl  (drops only ~2^-22 Al*Wl term).

__device__ __forceinline__ float4 relu4(float4 v) {
  v.x = fmaxf(v.x, 0.0f); v.y = fmaxf(v.y, 0.0f);
  v.z = fmaxf(v.z, 0.0f); v.w = fmaxf(v.w, 0.0f);
  return v;
}

__device__ __forceinline__ void split4(float4 v, v16h& ah, v16h& al, int base) {
  float f[4] = { v.x, v.y, v.z, v.w };
#pragma unroll
  for (int j = 0; j < 4; ++j) {
    _Float16 hi = (_Float16)f[j];
    ah[base + j] = hi;
    al[base + j] = (_Float16)(f[j] - (float)hi);
  }
}

__global__ __launch_bounds__(256)
void gemm_wmma_kernel(const float* __restrict__ A,
                      const _Float16* __restrict__ Wh, const _Float16* __restrict__ Wl,
                      float* __restrict__ out, int n_rows, int relu_in) {
  const int wave = threadIdx.x >> 5;
  const int lane = threadIdx.x & 31;
  const int lr   = lane & 15;
  const int sel  = lane >> 4;          // half-wave select
  const int r0   = blockIdx.x << 4;    // 16 rows per block
  const int col0 = wave << 4;          // 16 cols per wave

  int ar = r0 + lr;
  if (ar >= n_rows) ar = n_rows - 1;   // clamp loads, keep EXEC all-ones for WMMA
  const float*    arow = A  + (size_t)ar * 128;
  const _Float16* bhr  = Wh + (size_t)(col0 + lr) * 128;
  const _Float16* blr  = Wl + (size_t)(col0 + lr) * 128;

  v8f c = {};
#pragma unroll
  for (int kk = 0; kk < 4; ++kk) {
    const int k0 = kk * 32;
    // A fragment (16x32, contiguous K order, not the ISA table order): lanes 0-15: K k0+0..15,
    // lanes 16-31: K k0+16..31; matches B's contiguous K order.
    float4 x0 = *(const float4*)(arow + k0 + sel * 16);
    float4 x1 = *(const float4*)(arow + k0 + sel * 16 + 4);
    float4 y0 = *(const float4*)(arow + k0 + sel * 16 + 8);
    float4 y1 = *(const float4*)(arow + k0 + sel * 16 + 12);
    if (relu_in) { x0 = relu4(x0); x1 = relu4(x1); y0 = relu4(y0); y1 = relu4(y1); }
    v16h ah, al;
    split4(x0, ah, al, 0);
    split4(x1, ah, al, 4);
    split4(y0, ah, al, 8);
    split4(y1, ah, al, 12);
    // B fragment (32x16): lanes 0-15 hold K k0+0..15, lanes 16-31 K k0+16..31,
    // N = lane%16; contiguous in W^T row.
    v16h bh = *(const v16h*)(bhr + k0 + sel * 16);
    v16h bl = *(const v16h*)(blr + k0 + sel * 16);
    c = __builtin_amdgcn_wmma_f32_16x16x32_f16(false, ah, false, bh, (short)0, c, false, false);
    c = __builtin_amdgcn_wmma_f32_16x16x32_f16(false, al, false, bh, (short)0, c, false, false);
    c = __builtin_amdgcn_wmma_f32_16x16x32_f16(false, ah, false, bl, (short)0, c, false, false);
  }
  // C/D layout: VGPR i, lanes 0-15 -> M=i, lanes 16-31 -> M=8+i; N = lane%16
#pragma unroll
  for (int i = 0; i < 8; ++i) {
    int rr = r0 + sel * 8 + i;
    if (rr < n_rows) out[(size_t)rr * 128 + col0 + lr] = c[i];
  }
}

// ---------------------------------------------------------------- aggregation

// agg[i][f] = bias[f] + dinv[i]^2 * xw[i][f]   (self-loop + bias)
__global__ __launch_bounds__(256)
void init_agg_kernel(const float* __restrict__ xw, const float* __restrict__ dinv,
                     const float* __restrict__ bias, float* __restrict__ agg, int n_nodes) {
  int gid  = blockIdx.x * blockDim.x + threadIdx.x;
  int node = gid >> 5;
  if (node >= n_nodes) return;
  int f = (gid & 31) * 4;
  float di = dinv[node];
  float cc = di * di;
  float4 v = *(const float4*)(xw + (size_t)node * 128 + f);
  float4 b = *(const float4*)(bias + f);
  float4 o = { b.x + cc * v.x, b.y + cc * v.y, b.z + cc * v.z, b.w + cc * v.w };
  *(float4*)(agg + (size_t)node * 128 + f) = o;
}

// agg[dst] += dinv[src]*dinv[dst] * xw[src]   — one wave per edge, float4 gather
__global__ __launch_bounds__(256)
void scatter_kernel(const int* __restrict__ src, const int* __restrict__ dst,
                    const float* __restrict__ dinv, const float* __restrict__ xw,
                    float* __restrict__ agg, int n_edges) {
  int gid = blockIdx.x * blockDim.x + threadIdx.x;
  int e = gid >> 5;
  if (e >= n_edges) return;
  int f = (gid & 31) * 4;
  int s = src[e], d = dst[e];
  float coef = dinv[s] * dinv[d];
  float4 v = *(const float4*)(xw + (size_t)s * 128 + f);
  float* dp = agg + (size_t)d * 128 + f;
  atomicAdd(dp + 0, coef * v.x);
  atomicAdd(dp + 1, coef * v.y);
  atomicAdd(dp + 2, coef * v.z);
  atomicAdd(dp + 3, coef * v.w);
}

// ---------------------------------------------------------------- pooling

__global__ __launch_bounds__(256)
void pool_accum_kernel(const float* __restrict__ x, const int* __restrict__ batch,
                       float* __restrict__ sums, float* __restrict__ cnts, int n_nodes) {
  int gid  = blockIdx.x * blockDim.x + threadIdx.x;
  int node = gid >> 5;
  if (node >= n_nodes) return;
  int lane = gid & 31;
  int f = lane * 4;
  int g = batch[node];
  float4 v = *(const float4*)(x + (size_t)node * 128 + f);
  float* sp = sums + (size_t)g * 128 + f;
  atomicAdd(sp + 0, v.x);
  atomicAdd(sp + 1, v.y);
  atomicAdd(sp + 2, v.z);
  atomicAdd(sp + 3, v.w);
  if (lane == 0) atomicAdd(&cnts[g], 1.0f);
}

// ---------------------------------------------------------------- MLP + LayerNorm (exact f32)

__global__ __launch_bounds__(128)
void mlp_ln_kernel(const float* __restrict__ sums, const float* __restrict__ cnts,
                   const float* __restrict__ Wm1, const float* __restrict__ bm1,
                   const float* __restrict__ Wm2, const float* __restrict__ bm2,
                   const float* __restrict__ lng, const float* __restrict__ lnb,
                   float* __restrict__ out) {
  __shared__ float sm[128], hm[128], rs[128], rq[128];
  const int g = blockIdx.x;
  const int t = threadIdx.x;
  float cnt = fmaxf(cnts[g], 1.0f);
  sm[t] = sums[(size_t)g * 128 + t] / cnt;
  __syncthreads();
  float acc = bm1[t];
#pragma unroll 8
  for (int k = 0; k < 128; ++k) acc += sm[k] * Wm1[k * 128 + t];
  hm[t] = fmaxf(acc, 0.0f);
  __syncthreads();
  float y = bm2[t];
#pragma unroll 8
  for (int k = 0; k < 128; ++k) y += hm[k] * Wm2[k * 128 + t];
  rs[t] = y;
  rq[t] = y * y;
  __syncthreads();
  for (int o = 64; o > 0; o >>= 1) {
    if (t < o) { rs[t] += rs[t + o]; rq[t] += rq[t + o]; }
    __syncthreads();
  }
  float mu  = rs[0] * (1.0f / 128.0f);
  float var = rq[0] * (1.0f / 128.0f) - mu * mu;
  out[(size_t)g * 128 + t] = (y - mu) * rsqrtf(var + 1e-5f) * lng[t] + lnb[t];
}

// ---------------------------------------------------------------- launch

extern "C" void kernel_launch(void* const* d_in, const int* in_sizes, int n_in,
                              void* d_out, int out_size, void* d_ws, size_t ws_size,
                              hipStream_t stream) {
  const float* x0    = (const float*)d_in[0];
  const int*   ei    = (const int*)d_in[1];
  const int*   batch = (const int*)d_in[2];
  const float* Wl3[3] = { (const float*)d_in[3], (const float*)d_in[5], (const float*)d_in[7] };
  const float* bl3[3] = { (const float*)d_in[4], (const float*)d_in[6], (const float*)d_in[8] };
  const float* Wm1 = (const float*)d_in[9],  *bm1 = (const float*)d_in[10];
  const float* Wm2 = (const float*)d_in[11], *bm2 = (const float*)d_in[12];
  const float* lng = (const float*)d_in[13], *lnb = (const float*)d_in[14];

  const int N = in_sizes[2];        // batch vector length = #nodes
  const int E = in_sizes[1] / 2;    // edge_index is [2, E]
  const int* src = ei;
  const int* dst = ei + E;

  // workspace layout
  char* base = (char*)d_ws;
  size_t off = 0;
  float* buf0 = (float*)(base + off); off = align_up(off + (size_t)N * 128 * 4, 256);
  float* buf1 = (float*)(base + off); off = align_up(off + (size_t)N * 128 * 4, 256);
  float* dinv = (float*)(base + off); off = align_up(off + (size_t)N * 4, 256);
  _Float16* Wh = (_Float16*)(base + off); off = align_up(off + (size_t)128 * 128 * 2, 256);
  _Float16* Wl = (_Float16*)(base + off); off = align_up(off + (size_t)128 * 128 * 2, 256);
  float* sums = (float*)(base + off); off = align_up(off + (size_t)(64 * 128 + 64) * 4, 256);
  float* cnts = sums + 64 * 128;
  (void)ws_size; (void)n_in; (void)out_size;

  const int TB = 256;

  // symmetric GCN normalization with self-loops: deg = 1 + indeg; dinv = rsqrt(deg)
  fill_kernel<<<div_up_i(N, TB), TB, 0, stream>>>(dinv, 1.0f, N);
  deg_accum_kernel<<<div_up_i(E, TB), TB, 0, stream>>>(dst, dinv, E);
  rsqrt_inplace_kernel<<<div_up_i(N, TB), TB, 0, stream>>>(dinv, N);

  const float* xin = x0;
  for (int l = 0; l < 3; ++l) {
    convert_w_kernel<<<64, TB, 0, stream>>>(Wl3[l], Wh, Wl);
    gemm_wmma_kernel<<<div_up_i(N, 16), TB, 0, stream>>>(xin, Wh, Wl, buf1, N, l > 0 ? 1 : 0);
    init_agg_kernel<<<div_up_i((long long)N * 32, TB), TB, 0, stream>>>(buf1, dinv, bl3[l], buf0, N);
    scatter_kernel<<<div_up_i((long long)E * 32, TB), TB, 0, stream>>>(src, dst, dinv, buf1, buf0, E);
    xin = buf0;  // relu folded into next GEMM's A read; last layer unactivated
  }

  // global mean pool
  fill_kernel<<<div_up_i(64 * 128 + 64, TB), TB, 0, stream>>>(sums, 0.0f, 64 * 128 + 64);
  pool_accum_kernel<<<div_up_i((long long)N * 32, TB), TB, 0, stream>>>(buf0, batch, sums, cnts, N);

  // reasoner MLP + LayerNorm
  mlp_ln_kernel<<<64, 128, 0, stream>>>(sums, cnts, Wm1, bm1, Wm2, bm2, lng, lnb, (float*)d_out);
}
